// GeometricAttention_74242804678986
// MI455X (gfx1250) — compile-verified, matrix operand order checked by reading, not run on hardware
//
#include <hip/hip_runtime.h>

// ---------------------------------------------------------------------------
// GeometricAttention for MI455X (gfx1250, wave32, WMMA bf16).
// Single-pass flash-style attention: z (134 MB) is streamed through LDS once
// (vectorized b128 loads + packed bf16 LDS stores); k/v/q tiles are moved
// with gfx1250 async global->LDS copies (ASYNCcnt + s_wait_asynccnt).
// All GEMMs use v_wmma_f32_16x16x32_bf16 (f32 accumulate).
// ---------------------------------------------------------------------------

#define NB 2
#define LB 512
#define FB 128
#define CB 64
#define HB 12
#define DB 16
#define INF_C 100000.0f
#define SQ29_C 0.47140452079103173f
#define SCALE_C 0.5773502691896258f

typedef __attribute__((ext_vector_type(16))) __bf16 v16bf;
typedef __attribute__((ext_vector_type(4)))  __bf16 v4bf;
typedef __attribute__((ext_vector_type(8)))  float  v8f;

static __device__ __forceinline__ v8f vzero8() {
  v8f z;
#pragma unroll
  for (int i = 0; i < 8; ++i) z[i] = 0.0f;
  return z;
}

static __device__ __forceinline__ v8f wmma_bf16(v16bf a, v16bf b, v8f c) {
  // v_wmma_f32_16x16x32_bf16  D = A(16x32) * B(32x16) + C
  return __builtin_amdgcn_wmma_f32_16x16x32_bf16(false, a, false, b, (short)0, c,
                                                 false, false);
}

// Async global->LDS 16B/lane copy (gfx1250, tracked by ASYNCcnt).
// dsoff: per-lane LDS byte offset; gsrc: per-lane global address.
static __device__ __forceinline__ void async_b128(unsigned dsoff, const void* gsrc) {
  asm volatile("global_load_async_to_lds_b128 %0, %1, off"
               :: "v"(dsoff), "v"(gsrc) : "memory");
}
static __device__ __forceinline__ void wait_async0() {
  asm volatile("s_wait_asynccnt 0x0" ::: "memory");
}

// A-fragment (16x32, 16-bit) from row-major storage [row][k], stride ld.
// ISA layout: lane L holds row M=L%16; K runs {kb..kb+7} U {16+kb..16+kb+7},
// kb = 8*(L/16); VGPR g holds K pair (2g,2g+1).
static __device__ __forceinline__ v16bf frag_rowmajor(const __bf16* base, int ld) {
  const int lane = threadIdx.x & 31;
  const int r = lane & 15;
  const int kb = (lane >> 4) << 3;
  const __bf16* p0 = base + r * ld + kb;
  v16bf f;
#pragma unroll
  for (int t = 0; t < 8; ++t) { f[t] = p0[t]; f[8 + t] = p0[16 + t]; }
  return f;
}

// Same, but only K<16 is valid (zero-pad K 16..31). Used for per-head D=16.
static __device__ __forceinline__ v16bf frag_rowmajor_k16(const __bf16* base, int ld) {
  const int lane = threadIdx.x & 31;
  const int r = lane & 15;
  const int kb = (lane >> 4) << 3;
  const __bf16* p0 = base + r * ld + kb;
  v16bf f;
#pragma unroll
  for (int t = 0; t < 8; ++t) { f[t] = p0[t]; f[8 + t] = (__bf16)0.0f; }
  return f;
}

// B-fragment (32x16) from k-major storage [k][n], stride ld (lane -> column n).
static __device__ __forceinline__ v16bf frag_kmajor(const __bf16* base, int ld) {
  const int lane = threadIdx.x & 31;
  const int n = lane & 15;
  const int kb = (lane >> 4) << 3;
  v16bf f;
#pragma unroll
  for (int t = 0; t < 8; ++t) {
    f[t]     = base[(kb + t) * ld + n];
    f[8 + t] = base[(16 + kb + t) * ld + n];
  }
  return f;
}

// ---------------------------------------------------------------------------
// k0: precision conversion / packing of x and weights to bf16.
// ---------------------------------------------------------------------------
__global__ void k0_convert(const float* __restrict__ x, const float* __restrict__ Wq,
                           const float* __restrict__ Wk, const float* __restrict__ Wv,
                           const float* __restrict__ Wpb, const float* __restrict__ Wout,
                           __bf16* x_bf, __bf16* wqkv_bf, __bf16* wpb_bf, __bf16* wout_bf) {
  const int tid = blockIdx.x * blockDim.x + threadIdx.x;
  const int stride = gridDim.x * blockDim.x;
  for (int e = tid; e < NB * LB * FB; e += stride) x_bf[e] = (__bf16)x[e];
  for (int e = tid; e < 576 * 128; e += stride) {
    int o = e >> 7, k = e & 127;
    float v = (o < 192) ? Wq[o * 128 + k]
            : (o < 384) ? Wk[(o - 192) * 128 + k]
                        : Wv[(o - 384) * 128 + k];
    wqkv_bf[e] = (__bf16)v;
  }
  for (int e = tid; e < 16 * 64; e += stride) {
    int h = e >> 6, c = e & 63;
    wpb_bf[e] = (h < HB) ? (__bf16)Wpb[h * 64 + c] : (__bf16)0.0f;
  }
  for (int e = tid; e < 128 * 1056; e += stride) {
    int o = e / 1056, k = e % 1056;
    wout_bf[e] = (k < 1044) ? (__bf16)Wout[o * 1044 + k] : (__bf16)0.0f;
  }
}

// ---------------------------------------------------------------------------
// k1: fused QKV projection GEMM: (1024 x 576) = x_bf(1024x128) @ Wqkv^T.
// One wave per 16x16 output tile, K=128 -> 4 WMMAs.
// ---------------------------------------------------------------------------
__global__ __launch_bounds__(32) void k1_qkv(const __bf16* __restrict__ x_bf,
                                             const __bf16* __restrict__ wqkv_bf,
                                             __bf16* __restrict__ qkv_bf) {
  const int m0 = blockIdx.x * 16;
  const int n0 = blockIdx.y * 16;
  v8f c = vzero8();
#pragma unroll
  for (int ks = 0; ks < 4; ++ks) {
    v16bf a = frag_rowmajor(x_bf + (size_t)m0 * 128 + ks * 32, 128);
    v16bf b = frag_rowmajor(wqkv_bf + (size_t)n0 * 128 + ks * 32, 128);
    c = wmma_bf16(a, b, c);
  }
  const int lane = threadIdx.x & 31;
  const int half8 = (lane >> 4) << 3;
  const int col = lane & 15;
#pragma unroll
  for (int r = 0; r < 8; ++r)
    qkv_bf[(size_t)(m0 + r + half8) * 576 + n0 + col] = (__bf16)c[r];
}

// ---------------------------------------------------------------------------
// k2: flash attention. Grid: 64 WGs = (n, i-tile of 16). 8 waves / WG.
// Streams z in 16 j-tiles of 32; online softmax per (i,h); fused WMMA
// accumulation of alpha@z (per-i), alpha@v and alpha@p (per-head).
// Dynamic LDS ~148 KB (of 320 KB per WGP).
// ---------------------------------------------------------------------------
// LDS byte offsets inside the dynamic segment (assumed to start at LDS 0:
// kernel has no static __shared__).
#define OFF_Z   0u        // 16*32*64 bf16 (65536 B)
#define OFF_K   65536u    // 32*192 bf16 (12288 B)
#define OFF_V   77824u    // 32*192 bf16
#define OFF_Q   90112u    // 16*192 bf16 (6144 B)
#define OFF_S   98304u    // 16*32*16 f32 (32768 B)
#define OFF_P   131072u   // 16*16*32 bf16 (16384 B)
#define OFF_D2  147456u   // 16*32 f32
#define OFF_PJ  149504u   // 32*16 bf16
#define OFF_SC  150528u   // 16*16 f32
#define OFF_PCI 151552u   // 16*3 f32
#define OFF_MI  151744u   // 16 int
#define OFF_MJ  151808u   // 32 int
#define OFF_BC  151936u   // 16 f32
#define K2_LDS  152064u

__global__ __launch_bounds__(256) void k2_attn(
    const float* __restrict__ z, const float* __restrict__ p_CB,
    const unsigned char* __restrict__ mask, const float* __restrict__ gamma_raw,
    const __bf16* __restrict__ qkv_bf, const __bf16* __restrict__ wpb_bf,
    float* __restrict__ feat_p2n, float* __restrict__ feat_node,
    float* __restrict__ aggr) {
  extern __shared__ unsigned char sm[];
  __bf16* z_lds  = (__bf16*)(sm + OFF_Z);
  __bf16* k_lds  = (__bf16*)(sm + OFF_K);
  __bf16* v_lds  = (__bf16*)(sm + OFF_V);
  __bf16* q_lds  = (__bf16*)(sm + OFF_Q);
  float*  S_lds  = (float*) (sm + OFF_S);
  __bf16* P_lds  = (__bf16*)(sm + OFF_P);
  float*  d2_lds = (float*) (sm + OFF_D2);
  __bf16* pj_lds = (__bf16*)(sm + OFF_PJ);
  float*  sc_lds = (float*) (sm + OFF_SC);
  float*  pCBi   = (float*) (sm + OFF_PCI);
  int*    maski  = (int*)   (sm + OFF_MI);
  int*    maskj  = (int*)   (sm + OFF_MJ);
  float*  bc_lds = (float*) (sm + OFF_BC);

  const int tid = threadIdx.x;
  const int w = tid >> 5;
  const int lane = tid & 31;
  const int half8 = (lane >> 4) << 3;
  const int n = blockIdx.x >> 5;
  const int i0 = (blockIdx.x & 31) << 4;

  // ---- prologue: q tile via async global->LDS (already bf16 in global) ----
  // 16 rows * 384 B = 384 chunks of 16 B.
  for (int q = tid; q < 384; q += 256) {
    int i = q / 24, d0 = (q % 24) * 8;
    async_b128(OFF_Q + (unsigned)q * 16,
               qkv_bf + ((size_t)n * LB + i0 + i) * 576 + d0);
  }
  if (tid < 48) { int i = tid / 3, c = tid % 3; pCBi[tid] = p_CB[((size_t)n * LB + i0 + i) * 3 + c]; }
  if (tid < 16) maski[tid] = mask[(size_t)n * LB + i0 + tid];
  if (tid < 16)
    bc_lds[tid] = (tid < HB)
        ? -(logf(1.0f + __expf(gamma_raw[tid])) * SQ29_C * 0.5f) : 0.0f;
  // P rows h=12..15 stay zero forever (garbage only feeds unstored acc rows).
  for (int e = tid; e < 16 * 4 * 32; e += 256) {
    int i = e >> 7, rem = e & 127;
    P_lds[(i * 16 + 12 + (rem >> 5)) * 32 + (rem & 31)] = (__bf16)0.0f;
  }

  v8f accz[2][4];           // per wave: i in {2w,2w+1}, 4 c-subtiles of 16
  v8f accv[2], accp[2];     // per wave: heads {w, w+8 (w<4)}
#pragma unroll
  for (int a = 0; a < 2; ++a) {
#pragma unroll
    for (int b = 0; b < 4; ++b) accz[a][b] = vzero8();
    accv[a] = vzero8(); accp[a] = vzero8();
  }
  float m_run = -3.0e38f, s_run = 0.0f;
  const int i_sm = tid / HB, h_sm = tid % HB;  // softmax ownership (tid<192)

  wait_async0();
  __syncthreads();

  for (int jt = 0; jt < LB / 32; ++jt) {
    const int jg0 = jt * 32;
    // ---- k/v tiles: async global->LDS, 16 B per lane per op ----
    // each tile: 32 rows * 384 B = 768 chunks; 3 ops/thread/tile.
    for (int q = tid; q < 768; q += 256) {
      int j = q / 24, d0 = (q % 24) * 8;
      size_t row = ((size_t)n * LB + jg0 + j) * 576;
      async_b128(OFF_K + (unsigned)q * 16, qkv_bf + row + 192 + d0);
      async_b128(OFF_V + (unsigned)q * 16, qkv_bf + row + 384 + d0);
    }
    // ---- z tile: vectorized f32x4 load -> packed bf16x4 LDS store ----
    for (int e4 = tid * 4; e4 < 16 * 32 * 64; e4 += 256 * 4) {
      int i = e4 >> 11, j = (e4 >> 6) & 31, c = e4 & 63;
      const float4 zv =
          *(const float4*)&z[(((size_t)n * LB + i0 + i) * LB + jg0 + j) * 64 + c];
      v4bf pk;
      pk[0] = (__bf16)zv.x; pk[1] = (__bf16)zv.y;
      pk[2] = (__bf16)zv.z; pk[3] = (__bf16)zv.w;
      *(v4bf*)&z_lds[e4] = pk;
    }
    if (jt + 1 < LB / 32) {  // prefetch next z tile (global_prefetch_b8)
      size_t base = (((size_t)n * LB + i0 + (tid >> 4)) * LB + (jg0 + 32) + ((tid & 15) * 2)) * 64;
      __builtin_prefetch(z + base, 0, 1);
      __builtin_prefetch(z + base + 64, 0, 1);
    }
    for (int e = tid; e < 512; e += 256) {  // d2[i][j]
      int i = e >> 5, j = e & 31;
      const float* pj = p_CB + ((size_t)n * LB + jg0 + j) * 3;
      float dx = pCBi[i * 3 + 0] - pj[0];
      float dy = pCBi[i * 3 + 1] - pj[1];
      float dz = pCBi[i * 3 + 2] - pj[2];
      d2_lds[e] = dx * dx + dy * dy + dz * dz;
    }
    for (int e = tid; e < 32 * 16; e += 256) {  // coordinate B-tile (pad 3->16)
      int j = e >> 4, c = e & 15;
      pj_lds[e] = (c < 3) ? (__bf16)p_CB[((size_t)n * LB + jg0 + j) * 3 + c]
                          : (__bf16)0.0f;
    }
    if (tid < 32) maskj[tid] = mask[(size_t)n * LB + jg0 + tid];
    wait_async0();
    __syncthreads();

    // ---- Phase A: S_pair = z(pairs x 64c) @ Wpb^T, rows = (i,j) pairs ----
#pragma unroll
    for (int qi = 0; qi < 4; ++qi) {
      const int mt = w * 4 + qi;    // 32 tiles of 16 pairs
      const int p0 = mt * 16;
      v8f c = vzero8();
#pragma unroll
      for (int ks = 0; ks < 2; ++ks) {
        v16bf a = frag_rowmajor(z_lds + p0 * 64 + ks * 32, 64);
        v16bf b = frag_rowmajor(wpb_bf + ks * 32, 64);
        c = wmma_bf16(a, b, c);
      }
      const int hcol = lane & 15;
#pragma unroll
      for (int r = 0; r < 8; ++r) {
        int pr = p0 + r + half8;          // pr = i*32 + j
        S_lds[pr * 16 + hcol] = c[r];
      }
    }
    __syncthreads();

    // ---- Phase B: S_node += q_h @ k_h^T (per head, K=16 padded) ----
#pragma unroll
    for (int qi = 0; qi < 3; ++qi) {
      const int t = w * 3 + qi;     // 24 tiles: head x j-half
      const int h = t >> 1, jh = t & 1;
      v16bf a = frag_rowmajor_k16(q_lds + h * 16, 192);
      v16bf b = frag_rowmajor_k16(k_lds + (jh * 16) * 192 + h * 16, 192);
      v8f c = wmma_bf16(a, b, vzero8());
      const int jj = jh * 16 + (lane & 15);
#pragma unroll
      for (int r = 0; r < 8; ++r) {
        int i = r + half8;
        S_lds[(i * 32 + jj) * 16 + h] += c[r];
      }
    }
    __syncthreads();

    // ---- Phase C: online softmax, one thread per (i,h) ----
    if (tid < 16 * HB) {
      const float bc = bc_lds[h_sm];
      float vals[32];
      float tmax = -3.0e38f;
#pragma unroll
      for (int j = 0; j < 32; ++j) {
        float s = (S_lds[(i_sm * 32 + j) * 16 + h_sm] + d2_lds[i_sm * 32 + j] * bc) * SCALE_C;
        if (!(maski[i_sm] && maskj[j])) s -= INF_C;
        vals[j] = s;
        tmax = fmaxf(tmax, s);
      }
      float mnew = fmaxf(m_run, tmax);
      float rsc = __expf(m_run - mnew);
      float ssum = 0.0f;
#pragma unroll
      for (int j = 0; j < 32; ++j) {
        float p = __expf(vals[j] - mnew);
        ssum += p;
        P_lds[(i_sm * 16 + h_sm) * 32 + j] = (__bf16)p;
      }
      s_run = s_run * rsc + ssum;
      m_run = mnew;
      sc_lds[i_sm * 16 + h_sm] = rsc;
    } else {
      int t2 = tid - 192;
      sc_lds[(t2 >> 2) * 16 + 12 + (t2 & 3)] = 1.0f;  // pad rows: no rescale
    }
    __syncthreads();

    // ---- Phase D: rescale + WMMA accumulate ----
#pragma unroll
    for (int ii = 0; ii < 2; ++ii) {            // alpha @ z, per i
      const int i = w * 2 + ii;
      float f[8];
#pragma unroll
      for (int r = 0; r < 8; ++r) f[r] = sc_lds[i * 16 + r + half8];
      v16bf a = frag_rowmajor(P_lds + i * 512, 32);  // rows h, K=j
#pragma unroll
      for (int cs = 0; cs < 4; ++cs) {
        v8f acc = accz[ii][cs];
#pragma unroll
        for (int r = 0; r < 8; ++r) acc[r] *= f[r];
        v16bf b = frag_kmajor(z_lds + i * 2048 + cs * 16, 64);  // k=j, n=c
        accz[ii][cs] = wmma_bf16(a, b, acc);
      }
    }
    const int nh = (w < 4) ? 2 : 1;
    for (int e = 0; e < nh; ++e) {              // alpha @ v, alpha @ p, per head
      const int h = w + e * 8;
      float f[8];
#pragma unroll
      for (int r = 0; r < 8; ++r) f[r] = sc_lds[(r + half8) * 16 + h];
      v16bf a = frag_rowmajor(P_lds + h * 32, 512);  // rows i, K=j
      {
        v8f acc = accv[e];
#pragma unroll
        for (int r = 0; r < 8; ++r) acc[r] *= f[r];
        v16bf b = frag_kmajor(v_lds + h * 16, 192);
        accv[e] = wmma_bf16(a, b, acc);
      }
      {
        v8f acc = accp[e];
#pragma unroll
        for (int r = 0; r < 8; ++r) acc[r] *= f[r];
        v16bf b = frag_kmajor(pj_lds, 16);
        accp[e] = wmma_bf16(a, b, acc);
      }
    }
    __syncthreads();
  }

  // ---- finalize: alpha normalization (1/s), row masking, stores ----
  if (tid < 16 * HB) {
    float fin = (maski[i_sm] && s_run > 0.0f) ? (1.0f / s_run) : 0.0f;
    sc_lds[i_sm * 16 + h_sm] = fin;
  } else {
    int t2 = tid - 192;
    sc_lds[(t2 >> 2) * 16 + 12 + (t2 & 3)] = 0.0f;
  }
  __syncthreads();

#pragma unroll
  for (int ii = 0; ii < 2; ++ii) {
    const int i = w * 2 + ii;
    float f[8];
#pragma unroll
    for (int r = 0; r < 8; ++r) f[r] = sc_lds[i * 16 + r + half8];
    const size_t row = (size_t)n * LB + i0 + i;
#pragma unroll
    for (int cs = 0; cs < 4; ++cs)
#pragma unroll
      for (int r = 0; r < 8; ++r) {
        int h = r + half8;
        if (h < HB)
          feat_p2n[row * (HB * CB) + h * CB + cs * 16 + (lane & 15)] = accz[ii][cs][r] * f[r];
      }
  }
  {
    const int nh = (w < 4) ? 2 : 1;
    for (int e = 0; e < nh; ++e) {
      const int h = w + e * 8;
      float f[8];
#pragma unroll
      for (int r = 0; r < 8; ++r) f[r] = sc_lds[(r + half8) * 16 + h];
#pragma unroll
      for (int r = 0; r < 8; ++r) {
        const int irow = r + half8;
        const size_t row = (size_t)n * LB + i0 + irow;
        feat_node[row * (HB * DB) + h * DB + (lane & 15)] = accv[e][r] * f[r];
        if ((lane & 15) < 3)
          aggr[(row * HB + h) * 3 + (lane & 15)] = accp[e][r] * f[r];
      }
    }
  }
}

// ---------------------------------------------------------------------------
// k3: spatial features + feat_all assembly (bf16, K padded 1044 -> 1056).
// ---------------------------------------------------------------------------
__global__ void k3_feat(const float* __restrict__ p2n, const float* __restrict__ nodef,
                        const float* __restrict__ aggr, const float* __restrict__ R,
                        const float* __restrict__ t, __bf16* __restrict__ feat_all) {
  const int tid = blockIdx.x * blockDim.x + threadIdx.x;
  const int stride = gridDim.x * blockDim.x;
  for (int e = tid; e < NB * LB * 960; e += stride) {
    int row = e / 960, col = e % 960;
    float v = (col < 768) ? p2n[(size_t)row * 768 + col]
                          : nodef[(size_t)row * 192 + (col - 768)];
    feat_all[(size_t)row * 1056 + col] = (__bf16)v;
  }
  for (int e = tid; e < NB * LB * HB; e += stride) {
    int row = e / HB, h = e % HB;
    const float* Rl = R + (size_t)row * 9;
    const float* tl = t + (size_t)row * 3;
    const float* ag = aggr + ((size_t)row * HB + h) * 3;
    float d0 = ag[0] - tl[0], d1 = ag[1] - tl[1], d2v = ag[2] - tl[2];
    // local[i] = sum_j R[j][i] * d[j]
    float l0 = Rl[0] * d0 + Rl[3] * d1 + Rl[6] * d2v;
    float l1 = Rl[1] * d0 + Rl[4] * d1 + Rl[7] * d2v;
    float l2 = Rl[2] * d0 + Rl[5] * d1 + Rl[8] * d2v;
    float dist = sqrtf(l0 * l0 + l1 * l1 + l2 * l2);
    float inv = 1.0f / (dist + 1.0e-4f);
    __bf16* dst = feat_all + (size_t)row * 1056;
    dst[960 + h * 3 + 0] = (__bf16)l0;
    dst[960 + h * 3 + 1] = (__bf16)l1;
    dst[960 + h * 3 + 2] = (__bf16)l2;
    dst[996 + h] = (__bf16)dist;
    dst[1008 + h * 3 + 0] = (__bf16)(l0 * inv);
    dst[1008 + h * 3 + 1] = (__bf16)(l1 * inv);
    dst[1008 + h * 3 + 2] = (__bf16)(l2 * inv);
  }
  for (int e = tid; e < NB * LB * 12; e += stride) {
    int row = e / 12, c = e % 12;
    feat_all[(size_t)row * 1056 + 1044 + c] = (__bf16)0.0f;
  }
}

// ---------------------------------------------------------------------------
// k4: output GEMM (1024 x 128, K=1056) fused with bias, mask, residual, LN.
// One WG per 16-row tile; 8 waves each own a 16-col subtile (33 WMMAs).
// ---------------------------------------------------------------------------
__global__ __launch_bounds__(256) void k4_out(
    const __bf16* __restrict__ feat_all, const __bf16* __restrict__ wout_bf,
    const float* __restrict__ bout, const float* __restrict__ x,
    const unsigned char* __restrict__ mask, const float* __restrict__ ln_w,
    const float* __restrict__ ln_b, float* __restrict__ out) {
  __shared__ __bf16 A_lds[16 * 1056];
  __shared__ float y_lds[16 * 128];
  __shared__ float ps[256], pq[256];
  __shared__ float mu_s[16], rs_s[16];
  const int tid = threadIdx.x;
  const int r0 = blockIdx.x * 16;
  for (int e = tid; e < 16 * 1056; e += 256)
    A_lds[e] = feat_all[(size_t)(r0 + e / 1056) * 1056 + e % 1056];
  __syncthreads();

  const int w = tid >> 5, lane = tid & 31;
  const int half8 = (lane >> 4) << 3, ncol = lane & 15;
  const int nc = w * 16;
  v8f c = vzero8();
  for (int ks = 0; ks < 33; ++ks) {
    v16bf a = frag_rowmajor(A_lds + ks * 32, 1056);
    v16bf b = frag_rowmajor(wout_bf + (size_t)nc * 1056 + ks * 32, 1056);
    c = wmma_bf16(a, b, c);
  }
#pragma unroll
  for (int r = 0; r < 8; ++r) {
    const int lr = r + half8;
    const int fr = r0 + lr;
    const int col = nc + ncol;
    float v = c[r] + bout[col];
    if (!mask[fr]) v = 0.0f;
    v += x[(size_t)fr * 128 + col];
    y_lds[lr * 128 + col] = v;
  }
  __syncthreads();
  {
    const int row = tid >> 4, seg = tid & 15;
    float s = 0.0f, q2 = 0.0f;
#pragma unroll
    for (int t = 0; t < 8; ++t) {
      float v = y_lds[row * 128 + seg * 8 + t];
      s += v; q2 += v * v;
    }
    ps[tid] = s; pq[tid] = q2;
  }
  __syncthreads();
  if (tid < 16) {
    float s = 0.0f, q2 = 0.0f;
    for (int t = 0; t < 16; ++t) { s += ps[tid * 16 + t]; q2 += pq[tid * 16 + t]; }
    float mu = s * (1.0f / 128.0f);
    float var = q2 * (1.0f / 128.0f) - mu * mu;
    mu_s[tid] = mu;
    rs_s[tid] = rsqrtf(var + 1.0e-5f);
  }
  __syncthreads();
  for (int e = tid; e < 2048; e += 256) {
    int row = e >> 7, col = e & 127;
    out[(size_t)(r0 + row) * 128 + col] =
        (y_lds[e] - mu_s[row]) * rs_s[row] * ln_w[col] + ln_b[col];
  }
}

// ---------------------------------------------------------------------------
// launch
// ---------------------------------------------------------------------------
extern "C" void kernel_launch(void* const* d_in, const int* in_sizes, int n_in,
                              void* d_out, int out_size, void* d_ws, size_t ws_size,
                              hipStream_t stream) {
  const float* R         = (const float*)d_in[0];
  const float* t         = (const float*)d_in[1];
  const float* p_CB      = (const float*)d_in[2];
  const float* x         = (const float*)d_in[3];
  const float* z         = (const float*)d_in[4];
  const unsigned char* mask = (const unsigned char*)d_in[5];
  const float* Wq        = (const float*)d_in[6];
  const float* Wk        = (const float*)d_in[7];
  const float* Wv        = (const float*)d_in[8];
  const float* Wpb       = (const float*)d_in[9];
  const float* gamma_raw = (const float*)d_in[10];
  const float* Wout      = (const float*)d_in[11];
  const float* bout      = (const float*)d_in[12];
  const float* ln_w      = (const float*)d_in[13];
  const float* ln_b      = (const float*)d_in[14];
  float* out = (float*)d_out;

  unsigned char* ws = (unsigned char*)d_ws;
  size_t off = 0;
  auto carve = [&](size_t bytes) {
    unsigned char* p = ws + off;
    off = (off + bytes + 255) & ~(size_t)255;
    return p;
  };
  __bf16* x_bf      = (__bf16*)carve((size_t)NB * LB * FB * 2);
  __bf16* wqkv_bf   = (__bf16*)carve((size_t)576 * 128 * 2);
  __bf16* wpb_bf    = (__bf16*)carve((size_t)16 * 64 * 2);
  __bf16* wout_bf   = (__bf16*)carve((size_t)128 * 1056 * 2);
  __bf16* qkv_bf    = (__bf16*)carve((size_t)NB * LB * 576 * 2);
  float*  feat_p2n  = (float*)carve((size_t)NB * LB * HB * CB * 4);
  float*  feat_node = (float*)carve((size_t)NB * LB * HB * DB * 4);
  float*  aggr      = (float*)carve((size_t)NB * LB * HB * 3 * 4);
  __bf16* feat_all  = (__bf16*)carve((size_t)NB * LB * 1056 * 2);
  (void)ws_size; (void)in_sizes; (void)n_in; (void)out_size;

  k0_convert<<<256, 256, 0, stream>>>(x, Wq, Wk, Wv, Wpb, Wout,
                                      x_bf, wqkv_bf, wpb_bf, wout_bf);
  k1_qkv<<<dim3((NB * LB) / 16, 576 / 16), 32, 0, stream>>>(x_bf, wqkv_bf, qkv_bf);
  k2_attn<<<NB * (LB / 16), 256, K2_LDS, stream>>>(z, p_CB, mask, gamma_raw,
                                                   qkv_bf, wpb_bf,
                                                   feat_p2n, feat_node, aggr);
  k3_feat<<<256, 256, 0, stream>>>(feat_p2n, feat_node, aggr, R, t, feat_all);
  k4_out<<<(NB * LB) / 16, 256, 0, stream>>>(feat_all, wout_bf, bout, x, mask,
                                             ln_w, ln_b, out);
}
